// RNN_20263655702953
// MI455X (gfx1250) — matrix operand order repaired and checked by reading, not compiled, not run on hardware
//
#include <hip/hip_runtime.h>

// LSTM recurrence for MI455X (gfx1250, wave32 WMMA).
// Reference slices hs[:, -1, :] => output depends only on BATCH row 511 at
// every timestep t. One wave computes the 16-row batch tile containing row
// 511 (WMMA M=16); per step: 32 x v_wmma_f32_16x16x32_f16 (4 hidden tiles x
// 4 gates x 2 K-chunks), hardware tanh activations, and a shuffle-reduced FC
// for out[t, 0:2]. W_hh fragments are loop-invariant and held in registers
// (gfx1250 supports up to 1024 VGPRs/wave; kernel is single-wave by design).
// No in-loop barriers: all LDS traffic is intra-wave and LDS completes
// in order within a wave (CDNA5 ISA, DScnt "in order").

typedef _Float16 v16h __attribute__((ext_vector_type(16)));
typedef _Float16 v8h  __attribute__((ext_vector_type(8)));
typedef float    v8f  __attribute__((ext_vector_type(8)));

#define SEQ_T   1024
#define BATCH_N 512
#define HID     64
#define NGATE   256   // 4*HID, gate order [i | f | g | o]
#define BTILE   16
#define BBASE   (BATCH_N - BTILE)   // rows 496..511; only row 511 is live

#if defined(__has_builtin)
#if __has_builtin(__builtin_amdgcn_tanhf)
#define HW_TANH 1
#endif
#endif

__device__ __forceinline__ float tanh_(float x) {
#if defined(HW_TANH)
  return __builtin_amdgcn_tanhf(x);          // v_tanh_f32 (gfx1250 trans op)
#else
  return 2.0f / (1.0f + __expf(-2.0f * x)) - 1.0f;
#endif
}
__device__ __forceinline__ float sig_(float x) {
#if defined(HW_TANH)
  return 0.5f + 0.5f * __builtin_amdgcn_tanhf(0.5f * x);
#else
  return 1.0f / (1.0f + __expf(-x));
#endif
}

// A fragment (16-bit 16x32, contiguous layout matching B): half-wave hi needs
// K {off..off+15}, off = 16*hi (+ K-chunk base).
__device__ __forceinline__ v16h ld_fragA(const _Float16* row, int off0) {
  v8h a = *(const v8h*)(row + off0);
  v8h b = *(const v8h*)(row + off0 + 8);
  return __builtin_shufflevector(a, b, 0,1,2,3,4,5,6,7,8,9,10,11,12,13,14,15);
}
// B fragment (16-bit 32x16, contiguous K order, same as ld_fragA):
// half-wave hi holds K = kbase + 16*hi + j, j = 0..15 contiguous.
__device__ __forceinline__ v16h ld_fragB(const _Float16* row, int off0) {
  v8h a = *(const v8h*)(row + off0);
  v8h b = *(const v8h*)(row + off0 + 8);
  return __builtin_shufflevector(a, b, 0,1,2,3,4,5,6,7,8,9,10,11,12,13,14,15);
}

__global__ __launch_bounds__(32) void lstm_wmma_kernel(
    const float* __restrict__ xg,   // (T, B, 2)
    const float* __restrict__ Wih,  // (256, 2)
    const float* __restrict__ Whh,  // (256, 64)
    const float* __restrict__ bih,  // (256,)
    const float* __restrict__ bhh,  // (256,)
    const float* __restrict__ Wfc,  // (2, 64)
    const float* __restrict__ bfc,  // (2,)
    float* __restrict__ out)        // (T, 2)
{
  __shared__ __align__(16) _Float16 s_whh[NGATE * HID]; // [n][k] staging
  __shared__ __align__(16) _Float16 s_h[BTILE * HID];   // [m][k], A source
  __shared__ __align__(32) float    s_x[2 * BTILE];     // x tile, current step

  const int lane = threadIdx.x;     // 0..31
  const int hi   = lane >> 4;       // half-wave: 0 or 1
  const int n16  = lane & 15;       // column within a 16-wide tile
  const int offA = 16 * hi;         // A K sub-block per half-wave
  const int offB = 16 * hi;         // B K sub-block per half-wave

  // ---- one-time setup -------------------------------------------------
  for (int idx = lane; idx < NGATE * HID; idx += 32)
    s_whh[idx] = (_Float16)Whh[idx];                // W_hh -> f16 LDS
  for (int idx = lane; idx < BTILE * HID; idx += 32)
    s_h[idx] = (_Float16)0.0f;                      // h0 = 0

  // Loop-invariant per-lane gate-column constants (register resident).
  float bias_r[16], wi0[16], wi1[16];
  #pragma unroll
  for (int nt = 0; nt < 16; ++nt) {
    const int ncol = nt * 16 + n16;
    bias_r[nt] = bih[ncol] + bhh[ncol];
    wi0[nt] = Wih[ncol * 2 + 0];
    wi1[nt] = Wih[ncol * 2 + 1];
  }
  // Per-lane W_fc constants for the shuffle-reduced FC over row 511.
  float wfc0[4], wfc1[4];
  #pragma unroll
  for (int ht = 0; ht < 4; ++ht) {
    wfc0[ht] = Wfc[0 * HID + ht * 16 + n16];
    wfc1[ht] = Wfc[1 * HID + ht * 16 + n16];
  }
  const float myb = bfc[lane & 1];

  // c state in registers for the whole sequence.
  v8f cfr[4];
  #pragma unroll
  for (int ht = 0; ht < 4; ++ht)
    #pragma unroll
    for (int r = 0; r < 8; ++r) cfr[ht][r] = 0.0f;

  __syncthreads();

  // Hoist all 32 loop-invariant W_hh B-fragments into registers
  // (16 tiles x 2 K-chunks x 8 VGPRs = 256 VGPRs; single-wave kernel).
  v16h bfr[16][2];
  #pragma unroll
  for (int nt = 0; nt < 16; ++nt) {
    const _Float16* brow = &s_whh[(nt * 16 + n16) * HID];
    bfr[nt][0] = ld_fragB(brow, offB + 0);
    bfr[nt][1] = ld_fragB(brow, offB + 32);
  }

  // ---- sequential recurrence (no in-loop barriers) --------------------
  #pragma unroll 1
  for (int t = 0; t < SEQ_T; ++t) {
    // Stage x[t] tile: 16 rows x 2 inputs = 32 contiguous floats.
    s_x[lane] = xg[t * (BATCH_N * 2) + BBASE * 2 + lane];
    if (t + 1 < SEQ_T)
      __builtin_prefetch(&xg[(t + 1) * (BATCH_N * 2) + BBASE * 2 + lane], 0, 0);

    // A fragments from previous h (row m = lane%16, K-chunks 0-31 / 32-63).
    const _Float16* arow = &s_h[n16 * HID];
    v16h a0 = ld_fragA(arow, offA + 0);
    v16h a1 = ld_fragA(arow, offA + 32);

    // x values for this half-wave's 8 batch rows (two v8f LDS loads).
    v8f xa = *(const v8f*)&s_x[16 * hi];
    v8f xb = *(const v8f*)&s_x[16 * hi + 8];
    float x0r[8], x1r[8];
    #pragma unroll
    for (int r = 0; r < 4; ++r) {
      x0r[r]     = xa[2 * r];  x1r[r]     = xa[2 * r + 1];
      x0r[r + 4] = xb[2 * r];  x1r[r + 4] = xb[2 * r + 1];
    }

    float h7[4];   // h_new of batch row m = 7 + 8*hi (row 511 lives in hi=1)
    #pragma unroll
    for (int ht = 0; ht < 4; ++ht) {           // hidden-column tile
      v8f acc[4];
      #pragma unroll
      for (int g = 0; g < 4; ++g) {            // gate i,f,g,o -> tile g*4+ht
        const int nt = g * 4 + ht;
        // C init = bias + x_t @ W_ih.T (K=2 folded into 2 FMAs per element).
        v8f c;
        #pragma unroll
        for (int r = 0; r < 8; ++r)
          c[r] = bias_r[nt] + x0r[r] * wi0[nt] + x1r[r] * wi1[nt];
        c = __builtin_amdgcn_wmma_f32_16x16x32_f16(false, a0, false, bfr[nt][0],
                                                   (short)0, c, false, false);
        c = __builtin_amdgcn_wmma_f32_16x16x32_f16(false, a1, false, bfr[nt][1],
                                                   (short)0, c, false, false);
        acc[g] = c;
      }
      // Elementwise LSTM cell update in C-fragment layout:
      // lane holds (m = r + 8*hi, hcol = ht*16 + n16).
      #pragma unroll
      for (int r = 0; r < 8; ++r) {
        const float iv = sig_(acc[0][r]);
        const float fv = sig_(acc[1][r]);
        const float gv = tanh_(acc[2][r]);
        const float ov = sig_(acc[3][r]);
        const float cn = fv * cfr[ht][r] + iv * gv;
        cfr[ht][r] = cn;
        const float hn = ov * tanh_(cn);
        s_h[(r + 8 * hi) * HID + ht * 16 + n16] = (_Float16)hn;
        if (r == 7) h7[ht] = hn;               // m = 15 on hi=1 -> row 511
      }
    }

    // FC for the live row: out[t, o] = b_fc[o] + h_511 . W_fc[o, :].
    // Lanes 16..31 hold h_511 spread over n16; butterfly-reduce the 16 lanes.
    float p0 = h7[0] * wfc0[0] + h7[1] * wfc0[1] + h7[2] * wfc0[2] + h7[3] * wfc0[3];
    float p1 = h7[0] * wfc1[0] + h7[1] * wfc1[1] + h7[2] * wfc1[2] + h7[3] * wfc1[3];
    #pragma unroll
    for (int d = 1; d <= 8; d <<= 1) {
      p0 += __shfl_xor(p0, d, 32);
      p1 += __shfl_xor(p1, d, 32);
    }
    if ((lane & 30) == 16)                    // lanes 16 (o=0) and 17 (o=1)
      out[t * 2 + (lane & 1)] = ((lane & 1) ? p1 : p0) + myb;
  }
}

extern "C" void kernel_launch(void* const* d_in, const int* in_sizes, int n_in,
                              void* d_out, int out_size, void* d_ws, size_t ws_size,
                              hipStream_t stream) {
  (void)in_sizes; (void)n_in; (void)d_ws; (void)ws_size; (void)out_size;
  const float* x   = (const float*)d_in[0];
  const float* Wih = (const float*)d_in[1];
  const float* Whh = (const float*)d_in[2];
  const float* bih = (const float*)d_in[3];
  const float* bhh = (const float*)d_in[4];
  const float* Wfc = (const float*)d_in[5];
  const float* bfc = (const float*)d_in[6];
  float* out = (float*)d_out;
  lstm_wmma_kernel<<<1, 32, 0, stream>>>(x, Wih, Whh, bih, bhh, Wfc, bfc, out);
}
